// AlignmentLoss_60902636257514
// MI455X (gfx1250) — compile-verified, matrix operand order checked by reading, not run on hardware
//
#include <hip/hip_runtime.h>
#include <math.h>

typedef __attribute__((ext_vector_type(16))) _Float16 v16h;
typedef __attribute__((ext_vector_type(8)))  float    v8f;

// Problem constants (from reference setup_inputs)
constexpr int B  = 8;
constexpr int H  = 16;
constexpr int Lq = 128;
constexpr int Lc = 4096;
constexpr int D  = 1024;
constexpr int R  = H * Lq;          // 2048 reduction rows per batch
constexpr int RB = 16;              // row blocks (128 rows each)
constexpr int ROWS_PER_BLK = R / RB; // 128
constexpr int TOPK = 5;

// ---------------------------------------------------------------------------
// Stage A: column-sum of attn rows via WMMA  (ones[16x32] @ W-tile[32x16])
// Each wave owns 16 columns x 128 rows; 4 accumulated v_wmma_f32_16x16x32_f16.
// Lane mapping for B-matrix (32x16 f16): lanes 0-15 hold K=0..15, lanes 16-31
// hold K=16..31, column n = lane&15. Column sums are K-permutation invariant,
// so each per-lane load j makes the wave fetch two contiguous 64B row segments.
// ---------------------------------------------------------------------------
__global__ __launch_bounds__(256)
void attn_colsum_wmma(const float* __restrict__ attn, float* __restrict__ partial) {
    const int lane = threadIdx.x & 31;
    const int wave = threadIdx.x >> 5;
    const int b    = blockIdx.z;
    const int rb   = blockIdx.y;                       // 0..15
    const int cb   = blockIdx.x * 128 + wave * 16;     // 16-column tile
    const int n    = lane & 15;
    const int half = lane >> 4;

    v16h ones;
#pragma unroll
    for (int i = 0; i < 16; ++i) ones[i] = (_Float16)1.0f;

    v8f acc = {};
    const size_t batch_off = (size_t)b * R * Lc;
    const size_t col       = (size_t)cb + n;

#pragma unroll
    for (int it = 0; it < 4; ++it) {
        const int rbase = rb * ROWS_PER_BLK + it * 32 + half * 16;
        const float* p  = attn + batch_off + (size_t)rbase * Lc + col;
        v16h bm;
#pragma unroll
        for (int j = 0; j < 16; ++j) {
            bm[j] = (_Float16)p[(size_t)j * Lc];
        }
        acc = __builtin_amdgcn_wmma_f32_16x16x32_f16(
            /*neg_a=*/false, ones, /*neg_b=*/false, bm,
            /*c_mod=*/(short)0, acc, /*reuse_a=*/false, /*reuse_b=*/false);
    }

    // D[m][n] identical for all m: lane<16 holds n=lane in acc[0]
    if (lane < 16) {
        partial[(size_t)(b * RB + rb) * Lc + cb + lane] = acc[0];
    }
}

// ---------------------------------------------------------------------------
// Stage B: reduce 16 row-block partials -> avg_attn [B][Lc], scale 1/2048
// ---------------------------------------------------------------------------
__global__ __launch_bounds__(256)
void reduce_partials(const float* __restrict__ partial, float* __restrict__ avg) {
    const int i = blockIdx.x * 256 + threadIdx.x;    // 0 .. B*Lc-1
    if (i < B * Lc) {
        const int b  = i >> 12;
        const int lc = i & (Lc - 1);
        float s = 0.0f;
#pragma unroll
        for (int rb = 0; rb < RB; ++rb)
            s += partial[(size_t)(b * RB + rb) * Lc + lc];
        avg[i] = s * (1.0f / (float)R);
    }
}

// ---------------------------------------------------------------------------
// Stage C: per-batch top-5 (iterative argmax, ties -> lowest index)
// ---------------------------------------------------------------------------
__global__ __launch_bounds__(256)
void topk_kernel(const float* __restrict__ avg, int* __restrict__ idx_out) {
    __shared__ float vals[Lc];
    __shared__ float redv[256];
    __shared__ int   redi[256];
    const int b = blockIdx.x, tid = threadIdx.x;

    for (int i = tid; i < Lc; i += 256) vals[i] = avg[(size_t)b * Lc + i];
    __syncthreads();

    for (int k = 0; k < TOPK; ++k) {
        float best = -INFINITY; int bi = Lc;
        for (int i = tid; i < Lc; i += 256) {
            const float v = vals[i];
            if (v > best) { best = v; bi = i; }
        }
        redv[tid] = best; redi[tid] = bi;
        __syncthreads();
        for (int s = 128; s > 0; s >>= 1) {
            if (tid < s) {
                const float ov = redv[tid + s]; const int oi = redi[tid + s];
                if (ov > redv[tid] || (ov == redv[tid] && oi < redi[tid])) {
                    redv[tid] = ov; redi[tid] = oi;
                }
            }
            __syncthreads();
        }
        if (tid == 0) {
            idx_out[b * TOPK + k] = redi[0];
            vals[redi[0]] = -INFINITY;
        }
        __syncthreads();
    }
}

// ---------------------------------------------------------------------------
// Stage D/E: per-batch q_vec mean, norms, dots, partial loss
// ---------------------------------------------------------------------------
__device__ __forceinline__ float block_reduce_sum(float v, float* red, int tid) {
    red[tid] = v;
    __syncthreads();
    for (int s = 128; s > 0; s >>= 1) {
        if (tid < s) red[tid] += red[tid + s];
        __syncthreads();
    }
    const float r = red[0];
    __syncthreads();
    return r;
}

__global__ __launch_bounds__(256)
void loss_kernel(const float* __restrict__ q, const float* __restrict__ ctx,
                 const int* __restrict__ topk, float* __restrict__ lparts) {
    __shared__ float qv[D];
    __shared__ float red[256];
    const int b = blockIdx.x, tid = threadIdx.x;

    // q_vec[d] = mean over Lq; accumulate squared-norm contribution
    float qn_local = 0.0f;
    for (int d = tid; d < D; d += 256) {
        float s = 0.0f;
        for (int l = 0; l < Lq; ++l)
            s += q[((size_t)b * Lq + l) * D + d];
        const float m = s * (1.0f / (float)Lq);
        qv[d] = m;
        qn_local += m * m;
    }
    const float qn2 = block_reduce_sum(qn_local, red, tid);
    const float qn  = fmaxf(sqrtf(qn2), 1e-8f);

    float loss_acc = 0.0f;   // meaningful on tid 0 only
    for (int k = 0; k < TOPK; ++k) {
        const int ci = topk[b * TOPK + k];
        const float* cp = ctx + ((size_t)b * Lc + ci) * D;
        float dl = 0.0f, cl = 0.0f;
        for (int d = tid; d < D; d += 256) {
            const float c = cp[d];
            dl += qv[d] * c;
            cl += c * c;
        }
        const float dot = block_reduce_sum(dl, red, tid);
        const float cn2 = block_reduce_sum(cl, red, tid);
        const float cn  = fmaxf(sqrtf(cn2), 1e-8f);
        if (tid == 0) loss_acc += 1.0f - dot / (qn * cn);
    }
    if (tid == 0) lparts[b] = loss_acc;
}

__global__ void final_kernel(const float* __restrict__ lparts, float* __restrict__ out) {
    if (threadIdx.x == 0) {
        float s = 0.0f;
        for (int i = 0; i < B; ++i) s += lparts[i];
        out[0] = s * (1.0f / (float)(B * TOPK));
    }
}

// ---------------------------------------------------------------------------
extern "C" void kernel_launch(void* const* d_in, const int* in_sizes, int n_in,
                              void* d_out, int out_size, void* d_ws, size_t ws_size,
                              hipStream_t stream) {
    const float* q    = (const float*)d_in[0];   // question_emb [8,128,1024]
    const float* ctx  = (const float*)d_in[1];   // context_emb  [8,4096,1024]
    const float* attn = (const float*)d_in[2];   // cross_attn   [8,16,128,4096]
    float* out = (float*)d_out;

    // workspace layout (floats): partial [B*RB*Lc] | avg [B*Lc] | topk(64 ints) | lparts[8]
    float* partial = (float*)d_ws;
    float* avg     = partial + (size_t)B * RB * Lc;
    int*   topk    = (int*)(avg + (size_t)B * Lc);
    float* lparts  = (float*)(topk + 64);

    attn_colsum_wmma<<<dim3(Lc / 128, RB, B), 256, 0, stream>>>(attn, partial);
    reduce_partials <<<dim3((B * Lc) / 256), 256, 0, stream>>>(partial, avg);
    topk_kernel     <<<dim3(B), 256, 0, stream>>>(avg, topk);
    loss_kernel     <<<dim3(B), 256, 0, stream>>>(q, ctx, topk, lparts);
    final_kernel    <<<1, 32, 0, stream>>>(lparts, out);
}
